// EnergyPhysicsLayer_44023414784153
// MI455X (gfx1250) — compile-verified, matrix operand order checked by reading, not run on hardware
//
#include <hip/hip_runtime.h>

typedef __attribute__((ext_vector_type(16))) __bf16 v16bf;
typedef __attribute__((ext_vector_type(8)))  __bf16 v8bf;
typedef __attribute__((ext_vector_type(8)))  float  v8f;

#define KC 31
#define KT 61
#define H  512
#define W  2048
#define NB 16
#define NCHUNK 3              // u in [0,96) per kernel row, 3 chunks of K=32
#define BROWS  32             // output rows per block (two 16-row WMMA tiles)
#define TILE_ROWS 62          // 32 output rows + 30 halo
#define TILE_COLS 208         // 128 output cols + u-halo
#define BFRAG_ELEMS (KC * NCHUNK * 32 * 16)            // 47616 bf16 = 95232 B
#define SMEM_ELEMS  (TILE_ROWS * TILE_COLS + BFRAG_ELEMS)
#define SMEM_BYTES  (SMEM_ELEMS * 2)                   // 121,024 B < 320 KB LDS

// ---------------------------------------------------------------------------
// Setup: evaluate Gaussian kernel from scalars, normalize, and emit the
// Toeplitz (shifted-kernel) B matrix pre-swizzled into WMMA B-fragment layout:
// bfrag[((c*3 + j)*32 + lane)*16 + e] = w[c][32*j + kmap(lane,e) - (lane&15)]
// where kmap(lane,e) = (e&7) + ((e>>3)<<4) + ((lane>>4)<<3)  (mirrors the
// 16-bit 16x32 fragment layout, CDNA5 ISA section 7.12.2).
// ---------------------------------------------------------------------------
__global__ void build_bfrag_kernel(const float* __restrict__ log_slowness,
                                   const float* __restrict__ log_sigma,
                                   __bf16* __restrict__ bfrag) {
    __shared__ float w[KC * KT];
    __shared__ float total;
    const int tid = threadIdx.x;
    if (tid == 0) total = 0.0f;
    __syncthreads();

    const float s     = __expf(log_slowness[0]);
    const float sigma = __expf(log_sigma[0]);

    float part = 0.0f;
    for (int i = tid; i < KC * KT; i += blockDim.x) {
        const int c = i / KT;
        const int t = i - c * KT;
        const float gc = (float)(c - (KC / 2));
        const float gt = (float)(t - (KT / 2));
        const float diff = gt - fabsf(gc) * s;
        const float q = diff / sigma;
        const float v = __expf(-0.5f * q * q);
        w[i] = v;
        part += v;
    }
    atomicAdd(&total, part);
    __syncthreads();

    const float inv = 1.0f / (total + 1e-9f);

    for (int i = tid; i < BFRAG_ELEMS; i += blockDim.x) {
        const int e    = i & 15;
        const int lane = (i >> 4) & 31;
        const int j    = (i >> 9) % NCHUNK;
        const int c    = i / (16 * 32 * NCHUNK);
        const int kk   = (e & 7) + ((e >> 3) << 4) + ((lane >> 4) << 3);
        const int n    = lane & 15;
        const int t    = j * 32 + kk - n;       // u - n
        const float val = (t >= 0 && t < KT) ? w[c * KT + t] * inv : 0.0f;
        bfrag[i] = (__bf16)val;
    }
}

// ---------------------------------------------------------------------------
// Main conv: each block -> 32 rows x 128 cols of output for one batch image.
// 8 waves; each wave computes two y-stacked 16x16 WMMA tiles so every B
// fragment (read from LDS) feeds two v_wmma ops. Input tile + full B table
// staged in dynamic LDS.
// ---------------------------------------------------------------------------
__global__ void __launch_bounds__(256)
conv_wmma_kernel(const float* __restrict__ src,
                 const __bf16* __restrict__ bfrag,
                 float* __restrict__ out) {
    extern __shared__ __align__(32) __bf16 smem[];
    __bf16* tile = smem;                              // TILE_ROWS * TILE_COLS
    __bf16* bsh  = smem + TILE_ROWS * TILE_COLS;      // BFRAG_ELEMS (32B-aligned)

    const int b    = blockIdx.z;
    const int y0   = blockIdx.y * BROWS;
    const int xblk = blockIdx.x * 128;
    const int tid  = threadIdx.x;

    // Stage the 95KB pre-swizzled B table into LDS with 16B vector copies.
    {
        const uint4* __restrict__ gsrc = reinterpret_cast<const uint4*>(bfrag);
        uint4* ldst = reinterpret_cast<uint4*>(bsh);
        for (int i = tid; i < BFRAG_ELEMS / 8; i += 256) ldst[i] = gsrc[i];
    }

    // Cooperative load of (y0-15 .. y0+46) x (xblk-30 .. xblk+177), zero-padded
    // out of bounds, converted f32 -> bf16 on the fly.
    const float* __restrict__ img = src + (size_t)b * H * W;
    for (int i = tid; i < TILE_ROWS * TILE_COLS; i += 256) {
        const int r  = i / TILE_COLS;
        const int cc = i - r * TILE_COLS;
        const int gr = y0 - (KC / 2) + r;
        const int gc = xblk - (KT / 2) + cc;
        float v = 0.0f;
        if (gr >= 0 && gr < H && gc >= 0 && gc < W) v = img[(size_t)gr * W + gc];
        tile[i] = (__bf16)v;
    }
    __syncthreads();

    const int wv      = tid >> 5;        // wave id 0..7 -> 16-col sub-tile
    const int lane    = tid & 31;
    const int halfsel = lane >> 4;       // 0 or 1 (K-half selector)
    const int m       = lane & 15;       // row within tile / column within D

    v8f acc0 = {};
    v8f acc1 = {};
    for (int c = 0; c < KC; ++c) {
        const __bf16* row0p = &tile[(m + c) * TILE_COLS];
        const __bf16* row1p = row0p + 16 * TILE_COLS;
        #pragma unroll
        for (int j = 0; j < NCHUNK; ++j) {
            // B fragment from LDS (shared by both y-tiles).
            const v16bf bm = *reinterpret_cast<const v16bf*>(
                bsh + (((c * NCHUNK + j) * 32 + lane) << 4));

            // A fragments: lane holds row m, K values {k0..k0+7, k0+16..k0+23},
            // k0 = halfsel*8 -> two contiguous 16B runs per tile (ds_load_b128).
            const int col0 = wv * 16 + j * 32 + halfsel * 8;
            const v8bf lo0 = *reinterpret_cast<const v8bf*>(row0p + col0);
            const v8bf hi0 = *reinterpret_cast<const v8bf*>(row0p + col0 + 16);
            const v8bf lo1 = *reinterpret_cast<const v8bf*>(row1p + col0);
            const v8bf hi1 = *reinterpret_cast<const v8bf*>(row1p + col0 + 16);
            v16bf a0, a1;
            #pragma unroll
            for (int e = 0; e < 8; ++e) {
                a0[e] = lo0[e]; a0[e + 8] = hi0[e];
                a1[e] = lo1[e]; a1[e + 8] = hi1[e];
            }

            acc0 = __builtin_amdgcn_wmma_f32_16x16x32_bf16(
                false, a0, false, bm, (short)0, acc0, false, false);
            acc1 = __builtin_amdgcn_wmma_f32_16x16x32_bf16(
                false, a1, false, bm, (short)0, acc1, false, false);
        }
    }

    // D layout: VGPR v holds row (v + halfsel*8), column = lane&15.
    const int ocol = xblk + wv * 16 + m;
    float* __restrict__ obase0 =
        out + (size_t)b * H * W + (size_t)(y0 + halfsel * 8) * W + ocol;
    float* __restrict__ obase1 = obase0 + (size_t)16 * W;
    #pragma unroll
    for (int v = 0; v < 8; ++v) {
        obase0[(size_t)v * W] = acc0[v];
        obase1[(size_t)v * W] = acc1[v];
    }
}

extern "C" void kernel_launch(void* const* d_in, const int* in_sizes, int n_in,
                              void* d_out, int out_size, void* d_ws, size_t ws_size,
                              hipStream_t stream) {
    const float* src          = (const float*)d_in[0];
    const float* log_slowness = (const float*)d_in[1];
    const float* log_sigma    = (const float*)d_in[2];
    float*       out          = (float*)d_out;
    __bf16*      bfrag        = (__bf16*)d_ws;   // needs 95,232 bytes

    build_bfrag_kernel<<<1, 256, 0, stream>>>(log_slowness, log_sigma, bfrag);

    dim3 grid(W / 128, H / BROWS, NB);
    conv_wmma_kernel<<<grid, 256, SMEM_BYTES, stream>>>(src, bfrag, out);
}
